// PrismDecoder_8065948582272
// MI455X (gfx1250) — matrix operand order repaired and checked by reading, not compiled, not run on hardware
//
#include <hip/hip_runtime.h>
#include <hip/hip_bf16.h>

// ---------------------------------------------------------------------------
// PrismDecoder for MI455X (gfx1250, wave32, WMMA).
//
// Pipeline: gather+mean -> 4x GEMM (bf16 WMMA, f32 accum) -> 3x3 procrustes
// -> prism transform + atomic segment-sum -> normalize.
//
// GEMM: each wave32 computes a 16x(16*NB) output tile with
// v_wmma_f32_16x16x32_bf16 (NB=8 -> 16x128). Weights are pre-packed into the
// exact per-lane B-fragment order so the inner loop is pure contiguous b128
// loads. Waves within a block share one n-group (identical B stream -> WGP$
// hits), walking consecutive 16-row m-tiles.
// ---------------------------------------------------------------------------

#define N_VERTS 100000
#define N_FACES 200000
#define CDIM    512

typedef __attribute__((ext_vector_type(16))) __bf16 v16bf;
typedef __attribute__((ext_vector_type(8)))  __bf16 v8bf;
typedef __attribute__((ext_vector_type(4)))  __bf16 v4bf;
typedef __attribute__((ext_vector_type(8)))  float  v8f;

// ---------------------------------------------------------------------------
// Weight packing: dst is a sequence of 32(K) x 16(N) fragments, 512 bf16 each,
// fragment (kb, nb) at index (kb * (CoPad/16) + nb) * 512.
// Within a fragment, element (lane, e) holds W[kb*32 + (e&7) + 16*(e>>3) + 8*(lane>>4)]
//                                            [nb*16 + (lane & 15)]
// matching the 16-bit B-matrix lane striping used by the GEMM loader.
// ---------------------------------------------------------------------------
__global__ void pack_weights(const float* __restrict__ W, __bf16* __restrict__ dst,
                             int K, int CoPad, int CoReal) {
    int idx = blockIdx.x * blockDim.x + threadIdx.x;
    int total = K * CoPad;
    if (idx >= total) return;
    int e    = idx & 15;
    int lane = (idx >> 4) & 31;
    int frag = idx >> 9;
    int nblks = CoPad >> 4;
    int kb = frag / nblks;
    int nb = frag - kb * nblks;
    int n = nb * 16 + (lane & 15);
    int k = kb * 32 + (e & 7) + ((e >> 3) << 4) + ((lane >> 4) << 3);
    float v = (n < CoReal) ? W[(size_t)k * CoReal + n] : 0.0f;
    dst[idx] = (__bf16)v;
}

// ---------------------------------------------------------------------------
// feat[f] = mean(x[faces[f][0..2]]) in bf16, row-major F x 512.
// One block per face, 128 threads x float4.
// ---------------------------------------------------------------------------
__global__ void __launch_bounds__(128)
gather_mean(const float* __restrict__ x, const int* __restrict__ faces,
            __bf16* __restrict__ feat) {
    int f = blockIdx.x;
    int c = threadIdx.x << 2;
    int i0 = faces[3 * f + 0];
    int i1 = faces[3 * f + 1];
    int i2 = faces[3 * f + 2];
    float4 a = *(const float4*)(x + (size_t)i0 * CDIM + c);
    float4 b = *(const float4*)(x + (size_t)i1 * CDIM + c);
    float4 d = *(const float4*)(x + (size_t)i2 * CDIM + c);
    const float s = 1.0f / 3.0f;
    v4bf o;
    o[0] = (__bf16)((a.x + b.x + d.x) * s);
    o[1] = (__bf16)((a.y + b.y + d.y) * s);
    o[2] = (__bf16)((a.z + b.z + d.z) * s);
    o[3] = (__bf16)((a.w + b.w + d.w) * s);
    *(v4bf*)(feat + (size_t)f * CDIM + c) = o;
}

// ---------------------------------------------------------------------------
// bf16 GEMM + bias + ReLU:  Out[F x Co] = relu(A[F x K] @ W + b)
// Wave tile: 16 rows x (16*NB) cols. K-loop step 32.
// A-fragment per ISA 16-bit A layout:
//   lanes 0-15 : M=lane,    elems 0-7 = K 0-7,  elems 8-15 = K 16-23
//   lanes 16-31: M=lane-16, elems 0-7 = K 8-15, elems 8-15 = K 24-31
// -> two contiguous b128 loads per lane (base = k0 + 8*(lane>>4), base+16).
// Block mapping: m-tile fastest -> all 8 waves of a block share one n-group,
// so the B fragment stream is identical across the block (WGP$ resident).
// ---------------------------------------------------------------------------
template <int NB>
__global__ void __launch_bounds__(256)
gemm_bf16(const __bf16* __restrict__ A, const __bf16* __restrict__ Wp,
          const float* __restrict__ bias, __bf16* __restrict__ Out,
          int Fv, int K, int Co) {
    const int lane = threadIdx.x & 31;
    const int wid  = (blockIdx.x * blockDim.x + threadIdx.x) >> 5;
    const int ngroups = Co / (16 * NB);
    const int mtiles  = Fv >> 4;
    if (wid >= mtiles * ngroups) return;   // wave-uniform: EXEC stays all-ones
    const int ng = wid / mtiles;           // block shares one n-group
    const int mt = wid - ng * mtiles;
    const int m0 = mt << 4;
    const int n0 = ng * (16 * NB);
    const int nblks = Co >> 4;
    const int nb0   = n0 >> 4;

    v8f acc[NB];
#pragma unroll
    for (int j = 0; j < NB; ++j) acc[j] = (v8f){};

    const int m = m0 + (lane & 15);
    const __bf16* arow = A + (size_t)m * K + ((lane >> 4) << 3);

    for (int k0 = 0; k0 < K; k0 += 32) {
        union { v16bf v; v8bf h[2]; } a;
        a.h[0] = *reinterpret_cast<const v8bf*>(arow + k0);
        a.h[1] = *reinterpret_cast<const v8bf*>(arow + k0 + 16);
        __builtin_prefetch(arow + k0 + 64, 0, 3);   // WGP-scope global_prefetch_b8

        const __bf16* wf = Wp + ((size_t)(k0 >> 5) * nblks + nb0) * 512 + lane * 16;
#pragma unroll
        for (int j = 0; j < NB; ++j) {
            v16bf b = *reinterpret_cast<const v16bf*>(wf + (size_t)j * 512);
            acc[j] = __builtin_amdgcn_wmma_f32_16x16x32_bf16(
                false, a.v, false, b, (short)0, acc[j], false, false);
        }
    }

    // C/D layout: VGPR r, lane l -> row m0 + r + 8*(l>>4), col n0 + (l&15)
    const int col   = n0 + (lane & 15);
    const int rbase = m0 + ((lane >> 4) << 3);
    float bv[NB];
#pragma unroll
    for (int j = 0; j < NB; ++j) bv[j] = bias[col + 16 * j];
#pragma unroll
    for (int r = 0; r < 8; ++r) {
        __bf16* orow = Out + (size_t)(rbase + r) * Co + col;
#pragma unroll
        for (int j = 0; j < NB; ++j) {
            float v = fmaxf(acc[j][r] + bv[j], 0.0f);
            orow[16 * j] = (__bf16)v;
        }
    }
}

// ---------------------------------------------------------------------------
// Head: out12[F x 12] = h3[F x 256] @ W4 + b4 (W4 packed to 16 cols, zero pad).
// One wave per 16 rows, single accumulator, f32 output.
// ---------------------------------------------------------------------------
__global__ void __launch_bounds__(256)
gemm_head(const __bf16* __restrict__ A, const __bf16* __restrict__ Wp,
          const float* __restrict__ bias, float* __restrict__ Out12,
          int Fv, int K) {
    const int lane = threadIdx.x & 31;
    const int wid  = (blockIdx.x * blockDim.x + threadIdx.x) >> 5;
    const int mtiles = Fv >> 4;
    if (wid >= mtiles) return;
    const int m0 = wid << 4;

    v8f acc = {};
    const int m = m0 + (lane & 15);
    const __bf16* arow = A + (size_t)m * K + ((lane >> 4) << 3);

    for (int k0 = 0; k0 < K; k0 += 32) {
        union { v16bf v; v8bf h[2]; } a;
        a.h[0] = *reinterpret_cast<const v8bf*>(arow + k0);
        a.h[1] = *reinterpret_cast<const v8bf*>(arow + k0 + 16);
        v16bf b = *reinterpret_cast<const v16bf*>(Wp + (size_t)(k0 >> 5) * 512 + lane * 16);
        acc = __builtin_amdgcn_wmma_f32_16x16x32_bf16(
            false, a.v, false, b, (short)0, acc, false, false);
    }

    const int col   = lane & 15;
    const int rbase = m0 + ((lane >> 4) << 3);
    if (col < 12) {
        float bv = bias[col];
#pragma unroll
        for (int r = 0; r < 8; ++r)
            Out12[(size_t)(rbase + r) * 12 + col] = acc[r] + bv;
    }
}

// ---------------------------------------------------------------------------
// Per-face: special procrustes (Jacobi eig of M^T M -> SVD -> R with det=+1),
// write rotations, transform prism, atomic segment-sum into features+counts.
// ---------------------------------------------------------------------------
__global__ void __launch_bounds__(256)
face_finalize(const float* __restrict__ out12, const float* __restrict__ verts,
              const int* __restrict__ faces, float* __restrict__ rot_out,
              float* __restrict__ trans_out, float* __restrict__ feat_sum,
              float* __restrict__ counts, int Fv) {
    int f = blockIdx.x * blockDim.x + threadIdx.x;
    if (f >= Fv) return;

    float M[3][3];
#pragma unroll
    for (int i = 0; i < 3; ++i)
#pragma unroll
        for (int j = 0; j < 3; ++j)
            M[i][j] = out12[(size_t)f * 12 + 3 * i + j];
    float t[3];
#pragma unroll
    for (int j = 0; j < 3; ++j) t[j] = out12[(size_t)f * 12 + 9 + j];

    // S = M^T M (symmetric)
    float S[3][3];
#pragma unroll
    for (int i = 0; i < 3; ++i)
#pragma unroll
        for (int j = 0; j < 3; ++j)
            S[i][j] = M[0][i] * M[0][j] + M[1][i] * M[1][j] + M[2][i] * M[2][j];

    float V[3][3] = {{1.f, 0.f, 0.f}, {0.f, 1.f, 0.f}, {0.f, 0.f, 1.f}};

    // Cyclic Jacobi, 6 sweeps over (0,1),(0,2),(1,2)
#pragma unroll
    for (int sweep = 0; sweep < 6; ++sweep) {
#pragma unroll
        for (int pair = 0; pair < 3; ++pair) {
            const int p = (pair == 2) ? 1 : 0;
            const int q = (pair == 0) ? 1 : 2;
            float apq = S[p][q];
            if (fabsf(apq) > 1e-20f) {
                float tau = (S[q][q] - S[p][p]) / (2.0f * apq);
                float tj  = copysignf(1.0f, tau) / (fabsf(tau) + sqrtf(1.0f + tau * tau));
                float c   = rsqrtf(1.0f + tj * tj);
                float s   = tj * c;
#pragma unroll
                for (int k = 0; k < 3; ++k) {
                    float skp = S[k][p], skq = S[k][q];
                    S[k][p] = c * skp - s * skq;
                    S[k][q] = s * skp + c * skq;
                }
#pragma unroll
                for (int k = 0; k < 3; ++k) {
                    float spk = S[p][k], sqk = S[q][k];
                    S[p][k] = c * spk - s * sqk;
                    S[q][k] = s * spk + c * sqk;
                }
#pragma unroll
                for (int k = 0; k < 3; ++k) {
                    float vkp = V[k][p], vkq = V[k][q];
                    V[k][p] = c * vkp - s * vkq;
                    V[k][q] = s * vkp + c * vkq;
                }
            }
        }
    }

    float eig[3] = {S[0][0], S[1][1], S[2][2]};
    // Sort descending (swap V columns accordingly) so D flips the smallest sv.
#pragma unroll
    for (int a = 0; a < 3; ++a) {
        const int i = (a == 2) ? 1 : 0;
        const int j = (a == 0) ? 1 : 2;
        if (eig[i] < eig[j]) {
            float e = eig[i]; eig[i] = eig[j]; eig[j] = e;
#pragma unroll
            for (int k = 0; k < 3; ++k) {
                float v = V[k][i]; V[k][i] = V[k][j]; V[k][j] = v;
            }
        }
    }

    float inv_sig[3];
#pragma unroll
    for (int i = 0; i < 3; ++i) inv_sig[i] = rsqrtf(fmaxf(eig[i], 1e-12f));

    float U[3][3];
#pragma unroll
    for (int j = 0; j < 3; ++j)
#pragma unroll
        for (int i = 0; i < 3; ++i)
            U[i][j] = (M[i][0] * V[0][j] + M[i][1] * V[1][j] + M[i][2] * V[2][j]) * inv_sig[j];

    float detM = M[0][0] * (M[1][1] * M[2][2] - M[1][2] * M[2][1])
               - M[0][1] * (M[1][0] * M[2][2] - M[1][2] * M[2][0])
               + M[0][2] * (M[1][0] * M[2][1] - M[1][1] * M[2][0]);
    float s3 = (detM < 0.0f) ? -1.0f : 1.0f;   // det(U V^T) sign

    float R[3][3];
#pragma unroll
    for (int i = 0; i < 3; ++i)
#pragma unroll
        for (int j = 0; j < 3; ++j)
            R[i][j] = U[i][0] * V[j][0] + U[i][1] * V[j][1] + s3 * U[i][2] * V[j][2];

#pragma unroll
    for (int i = 0; i < 3; ++i)
#pragma unroll
        for (int j = 0; j < 3; ++j)
            rot_out[(size_t)f * 9 + 3 * i + j] = R[i][j];

    // Transform prism verts and scatter
#pragma unroll
    for (int a = 0; a < 3; ++a) {
        int idx = faces[3 * f + a];
        float p0 = verts[(size_t)idx * 3 + 0];
        float p1 = verts[(size_t)idx * 3 + 1];
        float p2 = verts[(size_t)idx * 3 + 2];
#pragma unroll
        for (int j = 0; j < 3; ++j) {
            float tp = p0 * R[0][j] + p1 * R[1][j] + p2 * R[2][j] + t[j];
            trans_out[(size_t)f * 9 + a * 3 + j] = tp;
            atomicAdd(&feat_sum[(size_t)idx * 3 + j], tp);
        }
        atomicAdd(&counts[idx], 1.0f);
    }
}

__global__ void normalize_feats(float* __restrict__ feat_sum,
                                const float* __restrict__ counts, int Nv) {
    int n = blockIdx.x * blockDim.x + threadIdx.x;
    if (n >= Nv) return;
    float inv = 1.0f / fmaxf(counts[n], 1.0f);
    feat_sum[(size_t)n * 3 + 0] *= inv;
    feat_sum[(size_t)n * 3 + 1] *= inv;
    feat_sum[(size_t)n * 3 + 2] *= inv;
}

// ---------------------------------------------------------------------------
extern "C" void kernel_launch(void* const* d_in, const int* in_sizes, int n_in,
                              void* d_out, int out_size, void* d_ws, size_t ws_size,
                              hipStream_t stream) {
    const float* x     = (const float*)d_in[0];
    const float* verts = (const float*)d_in[1];
    const int*   faces = (const int*)d_in[2];
    const float* W1 = (const float*)d_in[3];
    const float* b1 = (const float*)d_in[4];
    const float* W2 = (const float*)d_in[5];
    const float* b2 = (const float*)d_in[6];
    const float* W3 = (const float*)d_in[7];
    const float* b3 = (const float*)d_in[8];
    const float* W4 = (const float*)d_in[9];
    const float* b4 = (const float*)d_in[10];

    const int F = N_FACES, N = N_VERTS;

    // d_out layout: features (1,N,3) | transformed_prism (F,3,3) | rotations (F,3,3)
    float* features = (float*)d_out;
    float* trans    = features + (size_t)N * 3;
    float* rots     = trans + (size_t)F * 9;

    // workspace carve-out (ping-pong activation buffers)
    size_t off = 0;
    auto take = [&](size_t bytes) -> char* {
        char* p = (char*)d_ws + off;
        off = (off + bytes + 255) & ~(size_t)255;
        return p;
    };
    __bf16* bufA  = (__bf16*)take((size_t)F * CDIM * 2);  // feat, then h2
    __bf16* bufB  = (__bf16*)take((size_t)F * CDIM * 2);  // h1, then h3
    float*  out12 = (float*)take((size_t)F * 12 * 4);
    __bf16* W1p   = (__bf16*)take((size_t)512 * 512 * 2);
    __bf16* W2p   = (__bf16*)take((size_t)512 * 512 * 2);
    __bf16* W3p   = (__bf16*)take((size_t)512 * 256 * 2);
    __bf16* W4p   = (__bf16*)take((size_t)256 * 16 * 2);
    float*  counts = (float*)take((size_t)N * 4);

    hipMemsetAsync(features, 0, (size_t)N * 3 * 4, stream);
    hipMemsetAsync(counts,   0, (size_t)N * 4, stream);

    // one-time weight packing into WMMA B-fragment order
    pack_weights<<<(512 * 512 + 255) / 256, 256, 0, stream>>>(W1, W1p, 512, 512, 512);
    pack_weights<<<(512 * 512 + 255) / 256, 256, 0, stream>>>(W2, W2p, 512, 512, 512);
    pack_weights<<<(512 * 256 + 255) / 256, 256, 0, stream>>>(W3, W3p, 512, 256, 256);
    pack_weights<<<(256 * 16  + 255) / 256, 256, 0, stream>>>(W4, W4p, 256, 16, 12);

    gather_mean<<<F, 128, 0, stream>>>(x, faces, bufA);

    // layer 1: feat(bufA) -> h1(bufB), 512x512, 16x128 wave tiles
    {
        int waves = (F / 16) * (512 / 128);
        gemm_bf16<8><<<(waves + 7) / 8, 256, 0, stream>>>(bufA, W1p, b1, bufB, F, 512, 512);
    }
    // layer 2: h1(bufB) -> h2(bufA), 512x512
    {
        int waves = (F / 16) * (512 / 128);
        gemm_bf16<8><<<(waves + 7) / 8, 256, 0, stream>>>(bufB, W2p, b2, bufA, F, 512, 512);
    }
    // layer 3: h2(bufA) -> h3(bufB), 512x256
    {
        int waves = (F / 16) * (256 / 128);
        gemm_bf16<8><<<(waves + 7) / 8, 256, 0, stream>>>(bufA, W3p, b3, bufB, F, 512, 256);
    }
    // head: h3(bufB) -> out12, 256x12
    {
        int waves = F / 16;
        gemm_head<<<(waves + 7) / 8, 256, 0, stream>>>(bufB, W4p, b4, out12, F, 256);
    }

    face_finalize<<<(F + 255) / 256, 256, 0, stream>>>(out12, verts, faces, rots, trans,
                                                       features, counts, F);
    normalize_feats<<<(N + 255) / 256, 256, 0, stream>>>(features, counts, N);
}
